// GNN_graphpred_42391327212259
// MI455X (gfx1250) — matrix operand order repaired and checked by reading, not compiled, not run on hardware
//
#include <hip/hip_runtime.h>

typedef __attribute__((ext_vector_type(16))) __bf16 v16bf;
typedef __attribute__((ext_vector_type(8)))  __bf16 v8bf;
typedef __attribute__((ext_vector_type(8)))  float  v8f;

__device__ __forceinline__ unsigned short f2bf(float f) {
  union { float f; unsigned u; } x; x.f = f;
  unsigned r = x.u + 0x7FFFu + ((x.u >> 16) & 1u);   // round-to-nearest-even
  return (unsigned short)(r >> 16);
}

// fast sigmoid: native v_exp_f32 + native v_rcp_f32 (inputs already bf16-quantized)
__device__ __forceinline__ float fast_sigmoid(float x) {
  return __builtin_amdgcn_rcpf(1.0f + __expf(-x));
}

__global__ void fill_zero_kernel(float* __restrict__ p, long long n) {
  long long i = (long long)blockIdx.x * blockDim.x + threadIdx.x;
  const long long stride = (long long)gridDim.x * blockDim.x;
  for (; i < n; i += stride) p[i] = 0.0f;
}

// W: f32 [F][64] row-major  ->  Wt: bf16 [64][KPAD] column-major, zero-padded K
__global__ void pack_weights_kernel(const float* __restrict__ W,
                                    unsigned short* __restrict__ Wt,
                                    int F, int KPAD) {
  int idx = blockIdx.x * blockDim.x + threadIdx.x;
  if (idx >= 64 * KPAD) return;
  int n = idx / KPAD;
  int k = idx - n * KPAD;
  float v = (k < F) ? W[k * 64 + n] : 0.0f;
  Wt[idx] = f2bf(v);
}

// One wave = one 16-edge M-tile. feat = [h[src] | h[dst] | ea | p[src]-p[dst] | 0-pad]
// score = sigmoid(feat @ W + b); scatter-add score rows into h_out[dst].
template <int CIN, int KPAD>
__global__ __launch_bounds__(128) void edge_wmma_kernel(
    const float* __restrict__ h_in, const float* __restrict__ p,
    const float* __restrict__ ea, const int* __restrict__ src,
    const int* __restrict__ dst, const unsigned short* __restrict__ Wt,
    const float* __restrict__ bias, float* __restrict__ h_out, int n_edges) {
  __shared__ unsigned short sWt[64 * KPAD];     // B: bf16 col-major [64][KPAD]
  __shared__ unsigned short sA[4][16 * KPAD];   // A per wave: [16 edges][KPAD]
  __shared__ int sDst[4][16];

  const int tid  = threadIdx.x;
  const int wv   = tid >> 5;
  const int lane = tid & 31;

  for (int i = tid; i < 64 * KPAD; i += 128) sWt[i] = Wt[i];

  const int tile = blockIdx.x * 4 + wv;
  const int eb   = tile * 16;

  // ---- stage A tile: lanes stride K -> coalesced gathers of h[src]/h[dst]
  for (int e = 0; e < 16; ++e) {
    int edge = eb + e;
    if (edge >= n_edges) edge = n_edges - 1;   // clamp; masked at scatter
    const int s = src[edge];
    const int d = dst[edge];
    if (lane == 0) sDst[wv][e] = d;
    for (int k = lane; k < KPAD; k += 32) {
      float v;
      if (k < CIN)              v = h_in[(long long)s * CIN + k];
      else if (k < 2 * CIN)     v = h_in[(long long)d * CIN + (k - CIN)];
      else if (k == 2 * CIN)    v = ea[edge];
      else if (k < 2 * CIN + 4) {
        const int j = k - (2 * CIN + 1);
        v = p[s * 3 + j] - p[d * 3 + j];
      } else v = 0.0f;
      sA[wv][e * KPAD + k] = f2bf(v);
    }
  }
  __syncthreads();

  const int n_lo = lane & 15;
  const int k_hi = lane >> 4;

  // C/D 16x16 f32 layout: VGPR r, lanes 0-15 -> M=r, lanes 16-31 -> M=r+8; N=lane%16
  v8f acc[4];
#pragma unroll
  for (int nt = 0; nt < 4; ++nt) {
    const float bv = bias[nt * 16 + n_lo];
#pragma unroll
    for (int r = 0; r < 8; ++r) acc[nt][r] = bv;
  }

  const unsigned short* wa = &sA[wv][0];
#pragma unroll
  for (int ks = 0; ks < KPAD / 32; ++ks) {
    // A 16x32 bf16 fragment: lane row M = lane%16;
    // lanes 0-15 hold K {0..7,16..23}, lanes 16-31 hold K {8..15,24..31}
    const int m  = n_lo;
    const int ka = ks * 32 + k_hi * 8;
    v8bf alo = *reinterpret_cast<const v8bf*>(&wa[m * KPAD + ka]);
    v8bf ahi = *reinterpret_cast<const v8bf*>(&wa[m * KPAD + ka + 16]);
    v16bf a;
#pragma unroll
    for (int i = 0; i < 8; ++i) { a[i] = alo[i]; a[i + 8] = ahi[i]; }

#pragma unroll
    for (int nt = 0; nt < 4; ++nt) {
      // B 32x16 bf16 fragment: column N = lane%16, K = ks*32 + (lane/16)*8 + {0..7,16..23}
      const int col = nt * 16 + n_lo;
      const int kb  = ks * 32 + k_hi * 8;
      v8bf blo = *reinterpret_cast<const v8bf*>(&sWt[col * KPAD + kb]);
      v8bf bhi = *reinterpret_cast<const v8bf*>(&sWt[col * KPAD + kb + 16]);
      v16bf b;
#pragma unroll
      for (int i = 0; i < 8; ++i) { b[i] = blo[i]; b[i + 8] = bhi[i]; }
      acc[nt] = __builtin_amdgcn_wmma_f32_16x16x32_bf16(
          false, a, false, b, (short)0, acc[nt], false, false);
    }
  }

  // ---- sigmoid + scatter-add (segment_sum over dst)
  if (eb + 16 <= n_edges) {
    // fast path: full tile, no per-element masking
#pragma unroll
    for (int nt = 0; nt < 4; ++nt) {
      const int col = nt * 16 + n_lo;
#pragma unroll
      for (int r = 0; r < 8; ++r) {
        const int m = r + 8 * k_hi;
        atomicAdd(&h_out[(long long)sDst[wv][m] * 64 + col],
                  fast_sigmoid(acc[nt][r]));
      }
    }
  } else {
    // ragged tail: mask invalid edges
#pragma unroll
    for (int nt = 0; nt < 4; ++nt) {
      const int col = nt * 16 + n_lo;
#pragma unroll
      for (int r = 0; r < 8; ++r) {
        const int m = r + 8 * k_hi;
        if (eb + m < n_edges) {
          atomicAdd(&h_out[(long long)sDst[wv][m] * 64 + col],
                    fast_sigmoid(acc[nt][r]));
        }
      }
    }
  }
}

__global__ void pool_kernel(const float* __restrict__ h, const int* __restrict__ batch,
                            float* __restrict__ sums, float* __restrict__ counts,
                            long long n_nodes) {
  long long idx = (long long)blockIdx.x * blockDim.x + threadIdx.x;
  if (idx >= n_nodes * 64) return;
  const int node = (int)(idx >> 6);
  const int k    = (int)(idx & 63);
  const int g    = batch[node];
  atomicAdd(&sums[g * 64 + k], h[idx]);
  if (k == 0) atomicAdd(&counts[g], 1.0f);
}

__global__ void finalize_kernel(const float* __restrict__ sums,
                                const float* __restrict__ counts,
                                const float* __restrict__ Wp,
                                const float* __restrict__ bp,
                                float* __restrict__ out, int G) {
  int g = blockIdx.x * blockDim.x + threadIdx.x;
  if (g >= G) return;
  const float inv = 1.0f / fmaxf(counts[g], 1.0f);
  float* out_pred = out;       // tuple order: (pred[G,1], graph_rep[G,64])
  float* out_rep  = out + G;
  float acc = 0.0f;
  for (int k = 0; k < 64; ++k) {
    const float r = sums[g * 64 + k] * inv;
    out_rep[g * 64 + k] = r;
    acc += r * Wp[k];
  }
  out_pred[g] = acc + bp[0];
}

extern "C" void kernel_launch(void* const* d_in, const int* in_sizes, int n_in,
                              void* d_out, int out_size, void* d_ws, size_t ws_size,
                              hipStream_t stream) {
  (void)n_in; (void)ws_size;
  const float* x   = (const float*)d_in[0];
  const float* p   = (const float*)d_in[1];
  const int*   ei  = (const int*)d_in[2];
  const float* ea  = (const float*)d_in[3];
  const int* batch = (const int*)d_in[4];
  const float* W0  = (const float*)d_in[5];
  const float* b0  = (const float*)d_in[6];
  const float* W1  = (const float*)d_in[7];
  const float* b1  = (const float*)d_in[8];
  const float* W2  = (const float*)d_in[9];
  const float* b2  = (const float*)d_in[10];
  const float* Wp  = (const float*)d_in[11];
  const float* bp  = (const float*)d_in[12];

  const int n_nodes = in_sizes[0] / 5;
  const int n_edges = in_sizes[2] / 2;
  const int G       = out_size / 65;      // pred[G] + rep[G*64]
  const int* src = ei;
  const int* dst = ei + n_edges;

  auto align256 = [](size_t v) { return (v + 255) & ~(size_t)255; };
  char* ws = (char*)d_ws;
  size_t off = 0;
  const size_t hbytes = (size_t)n_nodes * 64 * sizeof(float);
  float* hA   = (float*)(ws + off); off = align256(off + hbytes);
  float* hB   = (float*)(ws + off); off = align256(off + hbytes);
  float* sums = (float*)(ws + off); off = align256(off + (size_t)G * 64 * sizeof(float));
  float* cnts = (float*)(ws + off); off = align256(off + (size_t)G * sizeof(float));
  unsigned short* Wt0 = (unsigned short*)(ws + off); off = align256(off + 64 * 32  * 2);
  unsigned short* Wt1 = (unsigned short*)(ws + off); off = align256(off + 64 * 160 * 2);
  unsigned short* Wt2 = (unsigned short*)(ws + off); off = align256(off + 64 * 160 * 2);

  // pack weights -> bf16 col-major, K padded to multiple of 32
  pack_weights_kernel<<<(64 * 32  + 255) / 256, 256, 0, stream>>>(W0, Wt0, 14,  32);
  pack_weights_kernel<<<(64 * 160 + 255) / 256, 256, 0, stream>>>(W1, Wt1, 132, 160);
  pack_weights_kernel<<<(64 * 160 + 255) / 256, 256, 0, stream>>>(W2, Wt2, 132, 160);

  const long long nh = (long long)n_nodes * 64;
  const int tiles   = (n_edges + 15) / 16;
  const int eblocks = (tiles + 3) / 4;

  fill_zero_kernel<<<2048, 256, 0, stream>>>(hA, nh);
  edge_wmma_kernel<5, 32><<<eblocks, 128, 0, stream>>>(x, p, ea, src, dst, Wt0, b0, hA, n_edges);

  fill_zero_kernel<<<2048, 256, 0, stream>>>(hB, nh);
  edge_wmma_kernel<64, 160><<<eblocks, 128, 0, stream>>>(hA, p, ea, src, dst, Wt1, b1, hB, n_edges);

  fill_zero_kernel<<<2048, 256, 0, stream>>>(hA, nh);
  edge_wmma_kernel<64, 160><<<eblocks, 128, 0, stream>>>(hB, p, ea, src, dst, Wt2, b2, hA, n_edges);

  fill_zero_kernel<<<64, 256, 0, stream>>>(sums, (long long)G * 64);
  fill_zero_kernel<<<4, 256, 0, stream>>>(cnts, (long long)G);
  pool_kernel<<<(int)((nh + 255) / 256), 256, 0, stream>>>(hA, batch, sums, cnts, n_nodes);

  finalize_kernel<<<(G + 255) / 256, 256, 0, stream>>>(sums, cnts, Wp, bp, (float*)d_out, G);
}
